// MultiHeadAttention_36515811950796
// MI455X (gfx1250) — compile-verified, matrix operand order checked by reading, not run on hardware
//
#include <hip/hip_runtime.h>
#include <hip/hip_bf16.h>
#include <math.h>

#define BATCH 4
#define SEQ   2048
#define DIM   1024
#define NH    16
#define HD    64
#define ROWS  (BATCH*SEQ)   // 8192

typedef __attribute__((ext_vector_type(16))) _Float16 v16h;
typedef __attribute__((ext_vector_type(4)))  _Float16 v4h;
typedef __attribute__((ext_vector_type(8)))  float    v8f;

union ABReg { v16h h; uint4 u[2]; _Float16 e[16]; };

__device__ __forceinline__ v8f wmma16(v16h a, v16h b, v8f c) {
    // D = A(16x32 f16) * B(32x16 f16) + C(16x16 f32)
    return __builtin_amdgcn_wmma_f32_16x16x32_f16(false, a, false, b, (short)0, c,
                                                  false, false);
}

// ---------------------------------------------------------------------------
// Weight transpose + f32->f16 convert:  Wt[n][k] = (f16) W[k][n]
// B-matrix lanes then read contiguous K-chunks from row n of Wt.
// ---------------------------------------------------------------------------
__global__ void __launch_bounds__(256)
wt_kernel(const float* __restrict__ W, _Float16* __restrict__ Wt) {
    __shared__ float tile[32][33];
    const int bx = blockIdx.x * 32, by = blockIdx.y * 32;
    const int tx = threadIdx.x, ty = threadIdx.y; // 32 x 8
    #pragma unroll
    for (int i = 0; i < 32; i += 8)
        tile[ty + i][tx] = W[(size_t)(by + ty + i) * DIM + bx + tx];
    __syncthreads();
    #pragma unroll
    for (int i = 0; i < 32; i += 8)
        Wt[(size_t)(bx + ty + i) * DIM + by + tx] = (_Float16)tile[tx][ty + i];
}

// ---------------------------------------------------------------------------
// Projection GEMM: Y = X(f32, ROWSxDIM) @ W + b, emitted as f16.
// mode 0/1: head-major (B,H,S,DK) for Q/K.  mode 2: transposed (B,H,DK,S) for V.
// One wave -> 16x64 output strip (4 WMMA accumulators), K-loop over DIM.
// ---------------------------------------------------------------------------
__global__ void __launch_bounds__(128)
proj_kernel(const float* __restrict__ X, const _Float16* __restrict__ Wt,
            const float* __restrict__ bias, _Float16* __restrict__ out, int mode) {
    const int lane = threadIdx.x & 31;
    const int wave = threadIdx.x >> 5;
    const int tileId = blockIdx.x * 4 + wave;
    const int m0 = (tileId >> 4) * 16;   // 512 row tiles
    const int n0 = (tileId & 15) * 64;   // 16 col groups
    const int lr = lane & 15;            // A-row / B-col inside tile
    const int hi = lane >> 4;
    const int g  = hi * 8;               // ISA half-chunk offset

    v8f c[4] = {};
    const float* xr0 = X + (size_t)(m0 + lr) * DIM;

    for (int kk = 0; kk < DIM; kk += 32) {
        float xf[16];
        const float* xr = xr0 + kk;
        *(float4*)(xf + 0)  = *(const float4*)(xr + g);
        *(float4*)(xf + 4)  = *(const float4*)(xr + g + 4);
        *(float4*)(xf + 8)  = *(const float4*)(xr + 16 + g);
        *(float4*)(xf + 12) = *(const float4*)(xr + 16 + g + 4);
        ABReg a;
        #pragma unroll
        for (int i = 0; i < 16; ++i) a.e[i] = (_Float16)xf[i];
        #pragma unroll
        for (int t = 0; t < 4; ++t) {
            const _Float16* wr = Wt + (size_t)(n0 + t * 16 + lr) * DIM + kk;
            ABReg b;
            b.u[0] = *(const uint4*)(wr + g);
            b.u[1] = *(const uint4*)(wr + 16 + g);
            c[t] = wmma16(a.h, b.h, c[t]);
        }
    }

    #pragma unroll
    for (int t = 0; t < 4; ++t) {
        const int cg = n0 + t * 16 + lr;
        const float bv = bias[cg];
        const int hh = cg >> 6, dk = cg & 63;
        #pragma unroll
        for (int vr = 0; vr < 8; ++vr) {
            const int r  = m0 + vr + 8 * hi;
            const int bb = r >> 11, ss = r & (SEQ - 1);
            const float val = c[t][vr] + bv;
            size_t addr;
            if (mode == 2) addr = ((size_t)(bb * NH + hh) * HD + dk) * SEQ + ss;
            else           addr = ((size_t)(bb * NH + hh) * SEQ + ss) * HD + dk;
            out[addr] = (_Float16)val;
        }
    }
}

// ---------------------------------------------------------------------------
// Fused attention: scores -> exact softmax (att to global) -> ctx = P @ V.
// One wave owns 16 query rows; its 16x2048 f16 score strip lives in LDS
// (64 KB/wave, 256 KB per 4-wave workgroup; WGP has 320 KB).
// ---------------------------------------------------------------------------
__global__ void __launch_bounds__(128)
attn_kernel(const _Float16* __restrict__ qh, const _Float16* __restrict__ kh,
            const _Float16* __restrict__ vt, float* __restrict__ att,
            _Float16* __restrict__ ctxh) {
    extern __shared__ _Float16 smem[];
    const int lane = threadIdx.x & 31;
    const int wave = threadIdx.x >> 5;
    const int bh = blockIdx.x >> 5;                  // b*NH + h
    const int q0 = (blockIdx.x & 31) * 64 + wave * 16;
    const int lr = lane & 15;
    const int hi = lane >> 4;
    const int g  = hi * 8;
    _Float16* P = smem + (size_t)wave * 16 * SEQ;

    // Q tile in A-layout, scale 1/sqrt(64) folded in.
    const _Float16* qrow = qh + ((size_t)bh * SEQ + q0 + lr) * HD;
    ABReg aq0, aq1;
    aq0.u[0] = *(const uint4*)(qrow + g);
    aq0.u[1] = *(const uint4*)(qrow + 16 + g);
    aq1.u[0] = *(const uint4*)(qrow + 32 + g);
    aq1.u[1] = *(const uint4*)(qrow + 48 + g);
    #pragma unroll
    for (int i = 0; i < 16; ++i) {
        aq0.e[i] *= (_Float16)0.125f;
        aq1.e[i] *= (_Float16)0.125f;
    }

    // ---- scores: 128 key tiles, 2 WMMAs each (DK=64 -> two K=32 steps) ----
    const _Float16* kbase = kh + (size_t)bh * SEQ * HD;
    for (int kt = 0; kt < SEQ / 16; ++kt) {
        const _Float16* krow = kbase + (size_t)(kt * 16 + lr) * HD;
        __builtin_prefetch(krow + 16 * HD, 0, 1);    // next key tile
        ABReg b0, b1;
        b0.u[0] = *(const uint4*)(krow + g);
        b0.u[1] = *(const uint4*)(krow + 16 + g);
        b1.u[0] = *(const uint4*)(krow + 32 + g);
        b1.u[1] = *(const uint4*)(krow + 48 + g);
        v8f c = {};
        c = wmma16(aq0.h, b0.h, c);
        c = wmma16(aq1.h, b1.h, c);
        #pragma unroll
        for (int vr = 0; vr < 8; ++vr)
            P[(size_t)(vr + 8 * hi) * SEQ + kt * 16 + lr] = (_Float16)c[vr];
    }
    __syncthreads();

    // ---- row max: lane lr handles row lr, half hi ----
    _Float16* pw = P + (size_t)lr * SEQ + hi * (SEQ / 2);
    float mx = -3.0e38f;
    for (int i = 0; i < SEQ / 2; i += 4) {
        v4h x = *(const v4h*)(pw + i);
        mx = fmaxf(mx, fmaxf(fmaxf((float)x[0], (float)x[1]),
                             fmaxf((float)x[2], (float)x[3])));
    }
    mx = fmaxf(mx, __shfl_xor(mx, 16, 32));

    // ---- exp + sum, unnormalized p back to LDS ----
    float sum = 0.f;
    for (int i = 0; i < SEQ / 2; i += 4) {
        v4h x = *(const v4h*)(pw + i);
        float e0 = __expf((float)x[0] - mx);
        float e1 = __expf((float)x[1] - mx);
        float e2 = __expf((float)x[2] - mx);
        float e3 = __expf((float)x[3] - mx);
        sum += (e0 + e1) + (e2 + e3);
        v4h y;
        y[0] = (_Float16)e0; y[1] = (_Float16)e1;
        y[2] = (_Float16)e2; y[3] = (_Float16)e3;
        *(v4h*)(pw + i) = y;
    }
    sum += __shfl_xor(sum, 16, 32);
    const float inv = 1.0f / sum;

    // ---- write normalized attention rows to d_out ----
    float* arow = att + ((size_t)bh * SEQ + q0 + lr) * SEQ + hi * (SEQ / 2);
    for (int i = 0; i < SEQ / 2; i += 4) {
        v4h x = *(const v4h*)(pw + i);
        float4 o;
        o.x = (float)x[0] * inv; o.y = (float)x[1] * inv;
        o.z = (float)x[2] * inv; o.w = (float)x[3] * inv;
        *(float4*)(arow + i) = o;
    }
    __syncthreads();

    // ---- ctx = softmax(P) @ V  (A from LDS, B from transposed V) ----
    v8f cc[4] = {};
    const _Float16 hinv = (_Float16)inv;   // lane's A-row is lr -> its own inv
    const _Float16* vbase = vt + (size_t)bh * HD * SEQ;
    for (int ks = 0; ks < SEQ / 32; ++ks) {
        ABReg pa;
        const _Float16* pl = P + (size_t)lr * SEQ + ks * 32;
        pa.u[0] = *(const uint4*)(pl + g);
        pa.u[1] = *(const uint4*)(pl + 16 + g);
        #pragma unroll
        for (int i = 0; i < 16; ++i) pa.e[i] *= hinv;
        #pragma unroll
        for (int dkt = 0; dkt < 4; ++dkt) {
            const _Float16* vrow = vbase + (size_t)(dkt * 16 + lr) * SEQ + ks * 32;
            ABReg bv;
            bv.u[0] = *(const uint4*)(vrow + g);
            bv.u[1] = *(const uint4*)(vrow + 16 + g);
            cc[dkt] = wmma16(pa.h, bv.h, cc[dkt]);
        }
    }

    // ---- store ctx as f16 (B,S,D) row-major for the output GEMM ----
    const int b = bh >> 4, h = bh & 15;
    #pragma unroll
    for (int dkt = 0; dkt < 4; ++dkt) {
        #pragma unroll
        for (int vr = 0; vr < 8; ++vr) {
            const int s = q0 + vr + 8 * hi;
            ctxh[(size_t)(b * SEQ + s) * DIM + h * HD + dkt * 16 + lr] =
                (_Float16)cc[dkt][vr];
        }
    }
}

// ---------------------------------------------------------------------------
// Output GEMM: out = ctx(f16) @ Wo + bo, f32 result straight to d_out.
// ---------------------------------------------------------------------------
__global__ void __launch_bounds__(128)
out_kernel(const _Float16* __restrict__ A, const _Float16* __restrict__ Wt,
           const float* __restrict__ bias, float* __restrict__ Y) {
    const int lane = threadIdx.x & 31;
    const int wave = threadIdx.x >> 5;
    const int tileId = blockIdx.x * 4 + wave;
    const int m0 = (tileId >> 4) * 16;
    const int n0 = (tileId & 15) * 64;
    const int lr = lane & 15;
    const int hi = lane >> 4;
    const int g  = hi * 8;

    v8f c[4] = {};
    const _Float16* ar0 = A + (size_t)(m0 + lr) * DIM;
    for (int kk = 0; kk < DIM; kk += 32) {
        ABReg a;
        a.u[0] = *(const uint4*)(ar0 + kk + g);
        a.u[1] = *(const uint4*)(ar0 + kk + 16 + g);
        #pragma unroll
        for (int t = 0; t < 4; ++t) {
            const _Float16* wr = Wt + (size_t)(n0 + t * 16 + lr) * DIM + kk;
            ABReg b;
            b.u[0] = *(const uint4*)(wr + g);
            b.u[1] = *(const uint4*)(wr + 16 + g);
            c[t] = wmma16(a.h, b.h, c[t]);
        }
    }
    #pragma unroll
    for (int t = 0; t < 4; ++t) {
        const int cg = n0 + t * 16 + lr;
        const float bv = bias[cg];
        #pragma unroll
        for (int vr = 0; vr < 8; ++vr) {
            const int r = m0 + vr + 8 * hi;
            Y[(size_t)r * DIM + cg] = c[t][vr] + bv;
        }
    }
}

// ---------------------------------------------------------------------------
extern "C" void kernel_launch(void* const* d_in, const int* in_sizes, int n_in,
                              void* d_out, int out_size, void* d_ws, size_t ws_size,
                              hipStream_t stream) {
    (void)in_sizes; (void)n_in; (void)out_size; (void)ws_size;
    const float* Q  = (const float*)d_in[0];
    const float* K  = (const float*)d_in[1];
    const float* V  = (const float*)d_in[2];
    const float* Wq = (const float*)d_in[3];
    const float* bq = (const float*)d_in[4];
    const float* Wk = (const float*)d_in[5];
    const float* bk = (const float*)d_in[6];
    const float* Wv = (const float*)d_in[7];
    const float* bv = (const float*)d_in[8];
    const float* Wo = (const float*)d_in[9];
    const float* bo = (const float*)d_in[10];

    float* out = (float*)d_out;
    float* att = out + (size_t)BATCH * SEQ * DIM;

    char* ws = (char*)d_ws;
    const size_t SZ_ACT = (size_t)BATCH * NH * SEQ * HD * sizeof(_Float16); // 16 MB
    _Float16* qh   = (_Float16*)(ws);
    _Float16* kh   = (_Float16*)(ws + SZ_ACT);
    _Float16* vth  = (_Float16*)(ws + 2 * SZ_ACT);
    _Float16* ctxh = (_Float16*)(ws + 3 * SZ_ACT);
    _Float16* wtq  = (_Float16*)(ws + 4 * SZ_ACT);
    _Float16* wtk  = wtq + (size_t)DIM * DIM;
    _Float16* wtv  = wtk + (size_t)DIM * DIM;
    _Float16* wto  = wtv + (size_t)DIM * DIM;

    dim3 tb(32, 8), tg(DIM / 32, DIM / 32);
    wt_kernel<<<tg, tb, 0, stream>>>(Wq, wtq);
    wt_kernel<<<tg, tb, 0, stream>>>(Wk, wtk);
    wt_kernel<<<tg, tb, 0, stream>>>(Wv, wtv);
    wt_kernel<<<tg, tb, 0, stream>>>(Wo, wto);

    const int gproj = (ROWS / 16) * (DIM / 64) / 4; // 2048 blocks of 4 waves
    proj_kernel<<<gproj, 128, 0, stream>>>(Q, wtq, bq, qh, 0);
    proj_kernel<<<gproj, 128, 0, stream>>>(K, wtk, bk, kh, 1);
    proj_kernel<<<gproj, 128, 0, stream>>>(V, wtv, bv, vth, 2);

    const size_t shbytes = (size_t)4 * 16 * SEQ * sizeof(_Float16); // 256 KB
    hipFuncSetAttribute((const void*)attn_kernel,
                        hipFuncAttributeMaxDynamicSharedMemorySize, (int)shbytes);
    attn_kernel<<<BATCH * NH * (SEQ / 64), 128, shbytes, stream>>>(qh, kh, vth,
                                                                   att, ctxh);

    out_kernel<<<gproj, 128, 0, stream>>>(ctxh, wto, bo, out);
}
